// ConditionalGraphGenerator_89129161326598
// MI455X (gfx1250) — compile-verified, matrix operand order checked by reading, not run on hardware
//
#include <hip/hip_runtime.h>

typedef _Float16 v16h __attribute__((ext_vector_type(16)));
typedef _Float16 v8h  __attribute__((ext_vector_type(8)));
typedef float    v8f  __attribute__((ext_vector_type(8)));
typedef int      v4i  __attribute__((ext_vector_type(4)));

constexpr int B = 256, N = 512, H = 128, LAT = 128, OUT = 2;

#define AS1 __attribute__((address_space(1)))
#define AS3 __attribute__((address_space(3)))

#if __has_builtin(__builtin_amdgcn_global_load_async_to_lds_b128)
#define USE_ASYNC_LDS 1
#else
#define USE_ASYNC_LDS 0
#endif

__device__ __forceinline__ void async_copy16(void* lds_dst, const void* gsrc) {
#if USE_ASYNC_LDS
    // param0: v4i addrspace(1)* (global src), param1: v4i addrspace(3)* (LDS dst)
    // generic LDS address low 32 bits == LDS offset (flat aperture rules)
    __builtin_amdgcn_global_load_async_to_lds_b128(
        (AS1 v4i*)(uintptr_t)gsrc,
        (AS3 v4i*)(uint32_t)(uintptr_t)lds_dst, 0, 0);
#else
    *(v8h*)lds_dst = *(const v8h*)gsrc;
#endif
}

__device__ __forceinline__ void wait_async_all() {
#if USE_ASYNC_LDS
#if __has_builtin(__builtin_amdgcn_s_wait_asynccnt)
    __builtin_amdgcn_s_wait_asynccnt(0);
#else
    asm volatile("s_wait_asynccnt 0x0" ::: "memory");
#endif
#endif
}

// ---------------------------------------------------------------- prep kernels

// w2t[n][k] = (f16) w_gcn2[k][n]   (transposed so B-fragments read contiguous k)
__global__ __launch_bounds__(256) void k_w2t(const float* __restrict__ w2,
                                             _Float16* __restrict__ w2t) {
    int idx = blockIdx.x * 256 + threadIdx.x;      // 0 .. 16383
    int n = idx >> 7, k = idx & 127;
    w2t[idx] = (_Float16)w2[k * H + n];
}

// ze = relu(z @ w_noise + b_noise); c2[b][o] = ze . w_out[H:,o] + b_out[o]
__global__ __launch_bounds__(128) void k_ze_c2(const float* __restrict__ z,
                                               const float* __restrict__ w_noise,
                                               const float* __restrict__ b_noise,
                                               const float* __restrict__ w_out,
                                               const float* __restrict__ b_out,
                                               float* __restrict__ c2) {
    __shared__ float red0[128], red1[128];
    int b = blockIdx.x, h = threadIdx.x;
    float s = b_noise[h];
    for (int l = 0; l < LAT; ++l) s += z[b * LAT + l] * w_noise[l * H + h];
    s = fmaxf(s, 0.0f);
    red0[h] = s * w_out[(H + h) * OUT + 0];
    red1[h] = s * w_out[(H + h) * OUT + 1];
    __syncthreads();
    for (int st = 64; st > 0; st >>= 1) {
        if (h < st) { red0[h] += red0[h + st]; red1[h] += red1[h + st]; }
        __syncthreads();
    }
    if (h < OUT) c2[b * OUT + h] = (h == 0 ? red0[0] : red1[0]) + b_out[h];
}

// X1_t[b][h][n] = (f16)(il[b][n][0]*w1[0][h] + il[b][n][1]*w1[1][h])   (no bias)
__global__ __launch_bounds__(256) void k_x1t(const float* __restrict__ il,
                                             const float* __restrict__ w1,
                                             _Float16* __restrict__ x1t) {
    int idx = blockIdx.x * 256 + threadIdx.x;      // B*H*N = 16M
    int n = idx & (N - 1);
    int h = (idx >> 9) & (H - 1);
    int b = idx >> 16;
    float2 xy = *(const float2*)(il + ((size_t)b * N + n) * 2);
    x1t[idx] = (_Float16)(xy.x * w1[h] + xy.y * w1[H + h]);
}

// one wave per adjacency row: dinv[b*N+r] = rsqrt(max(deg,1))
__global__ __launch_bounds__(256) void k_dinv(const float* __restrict__ adj,
                                              const int* __restrict__ num_nodes,
                                              float* __restrict__ dinv) {
    int wid = threadIdx.x >> 5, lane = threadIdx.x & 31;
    int row = blockIdx.x * 8 + wid;                // B*N rows
    int b = row >> 9, r = row & (N - 1);
    int nn = num_nodes[b];
    const float* ap = adj + (size_t)row * N;
    float s = 0.0f;
    for (int c = lane; c < N; c += 32) s += (c < nn) ? ap[c] : 0.0f;
    for (int off = 16; off > 0; off >>= 1) s += __shfl_xor(s, off, 32);
    float deg = (r < nn) ? (s + 1.0f) : 0.0f;
    if (lane == 0) dinv[row] = rsqrtf(fmaxf(deg, 1.0f));
}

// adj_n (f16) = dinv[r] * (mask*adj + eye*mask) * dinv[c]; 4 elems/thread
__global__ __launch_bounds__(256) void k_adjn(const float* __restrict__ adj,
                                              const float* __restrict__ dinv,
                                              const int* __restrict__ num_nodes,
                                              _Float16* __restrict__ adjn) {
    size_t idx = (size_t)blockIdx.x * 256 + threadIdx.x;   // B*N*N/4
    size_t e = idx * 4;
    int b = (int)(e >> 18);                 // N*N = 2^18
    int rem = (int)(e & ((N * N) - 1));
    int r = rem >> 9, c0 = rem & (N - 1);
    int nn = num_nodes[b];
    float4 a = *(const float4*)(adj + e);
    float4 dc = *(const float4*)(dinv + b * N + c0);
    float dr = dinv[b * N + r];
    bool rv = (r < nn);
    float av[4] = {a.x, a.y, a.z, a.w};
    float dv[4] = {dc.x, dc.y, dc.z, dc.w};
    _Float16 o[4];
    #pragma unroll
    for (int j = 0; j < 4; ++j) {
        int c = c0 + j;
        float v = (rv && c < nn) ? av[j] : 0.0f;
        if (c == r && rv) v += 1.0f;
        o[j] = (_Float16)(v * dr * dv[j]);
    }
    *(uint2*)(adjn + e) = *(uint2*)o;       // 8-byte store of 4 halves
}

// ---------------------------------------------------------------- WMMA GEMM
// C[b] (512 x 128) = A[b] (512 x KDIM, row-major) @ B (KDIM x 128)
// B supplied transposed: Bt[n][k] (so per-lane fragments read contiguous k).
// Workgroup: 256 threads / 8 waves computes a 128x128 C-tile.
// Double-buffered LDS, filled with GLOBAL_LOAD_ASYNC_TO_LDS_B128 (ASYNCcnt).

__device__ __forceinline__ v16h ld_frag(const _Float16* rowp, int khalf) {
    v8h lo = *(const v8h*)(rowp + khalf);
    v8h hi = *(const v8h*)(rowp + 16 + khalf);
    v16h r;
    #pragma unroll
    for (int i = 0; i < 8; ++i) { r[i] = lo[i]; r[i + 8] = hi[i]; }
    return r;
}

template <int KDIM, bool BIAS_RELU, bool STORE_T>
__global__ __launch_bounds__(256)
void k_gemm(const _Float16* __restrict__ A, int aPitch, size_t aBatch,
            const _Float16* __restrict__ Bt, int bPitch, size_t bBatch,
            const float* __restrict__ bias,
            _Float16* __restrict__ D, int dPitch, size_t dBatch) {
    __shared__ __align__(16) _Float16 As[2][128][40];  // pitch 40 halves = 20 banks
    __shared__ __align__(16) _Float16 Bs[2][128][40];  // -> conflict-free frag reads

    const int b = blockIdx.y;
    const int rowBase = blockIdx.x * 128;
    const _Float16* Ab = A + (size_t)b * aBatch + (size_t)rowBase * aPitch;
    const _Float16* Bb = Bt + (size_t)b * bBatch;

    const int t = threadIdx.x;
    const int lane = t & 31;
    const int wid = t >> 5;
    const int waveRow = wid >> 1;         // 0..3 -> 32-row strip
    const int waveCol = wid & 1;          // 0..1 -> 64-col strip
    const int lrow = lane & 15;
    const int khalf = (lane >> 4) * 8;    // ISA 16-bit A/B fragment layout

    // each thread stages two 16B chunks per tile: rows r0 and r0+64
    const int r0 = t >> 2;
    const int cc = (t & 3) * 8;

    v8f acc[2][4];
    #pragma unroll
    for (int i = 0; i < 2; ++i)
        #pragma unroll
        for (int j = 0; j < 4; ++j)
            acc[i][j] = (v8f){0, 0, 0, 0, 0, 0, 0, 0};

    auto stage = [&](int buf, int kk) {
        async_copy16(&As[buf][r0][cc],      Ab + (size_t)r0 * aPitch + kk + cc);
        async_copy16(&As[buf][r0 + 64][cc], Ab + (size_t)(r0 + 64) * aPitch + kk + cc);
        async_copy16(&Bs[buf][r0][cc],      Bb + (size_t)r0 * bPitch + kk + cc);
        async_copy16(&Bs[buf][r0 + 64][cc], Bb + (size_t)(r0 + 64) * bPitch + kk + cc);
    };

    stage(0, 0);
    wait_async_all();
    __syncthreads();

    int buf = 0;
    for (int kk = 0; kk < KDIM; kk += 32) {
        if (kk + 32 < KDIM) stage(buf ^ 1, kk + 32);   // prefetch next tile

        v16h af[2], bf[4];
        #pragma unroll
        for (int i = 0; i < 2; ++i)
            af[i] = ld_frag(&As[buf][waveRow * 32 + i * 16 + lrow][0], khalf);
        #pragma unroll
        for (int j = 0; j < 4; ++j)
            bf[j] = ld_frag(&Bs[buf][waveCol * 64 + j * 16 + lrow][0], khalf);

        #pragma unroll
        for (int i = 0; i < 2; ++i)
            #pragma unroll
            for (int j = 0; j < 4; ++j)
                acc[i][j] = __builtin_amdgcn_wmma_f32_16x16x32_f16(
                    false, af[i], false, bf[j], (short)0, acc[i][j], false, false);

        wait_async_all();       // next-tile LDS writes landed
        __syncthreads();        // waits this wave's ds reads, then barrier
        buf ^= 1;
    }

    // epilogue: lane L, vgpr v holds C[m0+v][n]  (m0 includes (L/16)*8)
    #pragma unroll
    for (int i = 0; i < 2; ++i) {
        int m0 = rowBase + waveRow * 32 + i * 16 + (lane >> 4) * 8;
        #pragma unroll
        for (int j = 0; j < 4; ++j) {
            int n = waveCol * 64 + j * 16 + lrow;
            if constexpr (BIAS_RELU) {
                float bv = bias[n];
                #pragma unroll
                for (int v = 0; v < 8; ++v) {
                    float f = acc[i][j][v] + bv;
                    f = f > 0.0f ? f : 0.0f;
                    D[(size_t)b * dBatch + (size_t)(m0 + v) * dPitch + n] = (_Float16)f;
                }
            } else {
                (void)bias;
                v8h o;
                #pragma unroll
                for (int v = 0; v < 8; ++v) o[v] = (_Float16)acc[i][j][v];
                if constexpr (STORE_T)   // store transposed: Dt[n][m0..m0+7]
                    *(v8h*)(D + (size_t)b * dBatch + (size_t)n * dPitch + m0) = o;
            }
        }
    }
}

// ---------------------------------------------------------------- output GEMV
__global__ __launch_bounds__(128) void k_out(const _Float16* __restrict__ h2,
                                             const float* __restrict__ w_out,
                                             const float* __restrict__ c2,
                                             const int* __restrict__ num_nodes,
                                             float* __restrict__ out) {
    __shared__ float wo[2 * H];
    wo[threadIdx.x] = w_out[threadIdx.x];
    wo[threadIdx.x + 128] = w_out[threadIdx.x + 128];
    __syncthreads();
    int idx = blockIdx.x * 128 + threadIdx.x;      // B*N
    int b = idx >> 9, n = idx & (N - 1);
    int nn = num_nodes[b];
    float s0 = 0.0f, s1 = 0.0f;
    if (n < nn) {
        s0 = c2[b * OUT + 0];
        s1 = c2[b * OUT + 1];
        const v8h* hp = (const v8h*)(h2 + (size_t)idx * H);
        for (int hh = 0; hh < H / 8; ++hh) {
            v8h v = hp[hh];
            #pragma unroll
            for (int j = 0; j < 8; ++j) {
                float f = (float)v[j];
                int h = hh * 8 + j;
                s0 += f * wo[h * 2 + 0];
                s1 += f * wo[h * 2 + 1];
            }
        }
    }
    out[(size_t)idx * 2 + 0] = s0;
    out[(size_t)idx * 2 + 1] = s1;
}

// ---------------------------------------------------------------- launch

extern "C" void kernel_launch(void* const* d_in, const int* in_sizes, int n_in,
                              void* d_out, int out_size, void* d_ws, size_t ws_size,
                              hipStream_t stream) {
    const float* z       = (const float*)d_in[0];
    const float* il      = (const float*)d_in[1];
    const float* adj     = (const float*)d_in[2];
    const int*   nn      = (const int*)  d_in[3];
    const float* w1      = (const float*)d_in[4];
    const float* b1      = (const float*)d_in[5];
    const float* w2      = (const float*)d_in[6];
    const float* b2      = (const float*)d_in[7];
    const float* w_noise = (const float*)d_in[8];
    const float* b_noise = (const float*)d_in[9];
    const float* w_out   = (const float*)d_in[10];
    const float* b_out   = (const float*)d_in[11];
    float* out = (float*)d_out;

    char* ws = (char*)d_ws;
    size_t off = 0;
    auto alloc = [&](size_t bytes) -> char* {
        char* p = ws + off;
        off = (off + bytes + 255) & ~(size_t)255;
        return p;
    };
    float*    dinv = (float*)   alloc((size_t)B * N * 4);
    float*    c2   = (float*)   alloc((size_t)B * OUT * 4);
    _Float16* w2t  = (_Float16*)alloc((size_t)H * H * 2);
    _Float16* x1t  = (_Float16*)alloc((size_t)B * H * N * 2);   // reused as HW_t
    _Float16* adjn = (_Float16*)alloc((size_t)B * N * N * 2);
    _Float16* h1   = (_Float16*)alloc((size_t)B * N * H * 2);   // reused as h2
    _Float16* hwt = x1t;   // X1_t dead after layer-1 GEMM
    _Float16* h2  = h1;    // h1 dead after h1@w2

    k_w2t  <<<H * H / 256, 256, 0, stream>>>(w2, w2t);
    k_ze_c2<<<B, 128, 0, stream>>>(z, w_noise, b_noise, w_out, b_out, c2);
    k_x1t  <<<(B * H * N) / 256, 256, 0, stream>>>(il, w1, x1t);
    k_dinv <<<B * N / 8, 256, 0, stream>>>(adj, nn, dinv);
    k_adjn <<<(size_t)B * N * N / 4 / 256, 256, 0, stream>>>(adj, dinv, nn, adjn);

    dim3 g(4, B);
    // layer 1: h1 = relu(adj_n @ X1 + b1)          [row-major f16 out]
    k_gemm<N, true, false><<<g, 256, 0, stream>>>(
        adjn, N, (size_t)N * N, x1t, N, (size_t)H * N, b1, h1, H, (size_t)N * H);
    // HW_t = (h1 @ w_gcn2)^T                        [transposed f16 out]
    k_gemm<H, false, true><<<g, 256, 0, stream>>>(
        h1, H, (size_t)N * H, w2t, H, (size_t)0, nullptr, hwt, N, (size_t)H * N);
    // layer 2: h2 = relu(adj_n @ HW + b2)
    k_gemm<N, true, false><<<g, 256, 0, stream>>>(
        adjn, N, (size_t)N * N, hwt, N, (size_t)H * N, b2, h2, H, (size_t)N * H);

    k_out<<<B * N / 128, 128, 0, stream>>>(h2, w_out, c2, nn, out);
}
